// Attention_50165218017954
// MI455X (gfx1250) — matrix operand order repaired and checked by reading, not compiled, not run on hardware
//
#include <hip/hip_runtime.h>
#include <hip/hip_bf16.h>

typedef __attribute__((ext_vector_type(16))) _Float16 v16h;
typedef __attribute__((ext_vector_type(8)))  float    v8f;

#define WMMA_F16(a, b, c) \
  __builtin_amdgcn_wmma_f32_16x16x32_f16(false, (a), false, (b), (short)0, (c), false, false)

// ---- CDNA5 async global->LDS copy (ASYNCcnt path), 16 bytes per lane -------
__device__ __forceinline__ void async_b128(void* lds_dst, const void* gsrc) {
  unsigned lds = (unsigned)(size_t)lds_dst;           // LDS addr = flat[31:0]
  unsigned long long ga = (unsigned long long)(size_t)gsrc;
  asm volatile("global_load_async_to_lds_b128 %0, %1, off"
               :: "v"(lds), "v"(ga)
               : "memory");
}
__device__ __forceinline__ void wait_async0() {
  asm volatile("s_wait_asynccnt 0x0" ::: "memory");
}

// ---------------------------------------------------------------- f32 -> f16
__global__ void cvt_f32_to_f16(const float* __restrict__ src,
                               _Float16* __restrict__ dst, long n) {
  long i = (long)blockIdx.x * blockDim.x + threadIdx.x;
  if (i < n) dst[i] = (_Float16)src[i];
}

// ------------------------------------------------------- tiled WMMA f16 GEMM
// C[M,Nc] (f32) = A[M,K] (f16, row-major) @ B[K,Nc] (f16, row-major) + bias
#define BM 128
#define BN 128
#define BK 32

__global__ __launch_bounds__(256) void gemm_f16_wmma(
    const _Float16* __restrict__ A, const _Float16* __restrict__ Bm,
    const float* __restrict__ bias, float* __restrict__ C,
    int M, int Nc, int K) {
  __shared__ __align__(16) _Float16 As[BM * BK];
  __shared__ __align__(16) _Float16 Bs[BK * BN];

  const int tid   = threadIdx.x;
  const int lane  = tid & 31;
  const int wave  = tid >> 5;
  const int waveM = wave & 3;   // 4 waves along M, 32 rows each
  const int waveN = wave >> 2;  // 2 waves along N, 64 cols each
  const int bm = blockIdx.y * BM;
  const int bn = blockIdx.x * BN;
  const int arow = lane & 15;
  const int kh   = lane >> 4;

  const int ra = tid >> 1, ca = (tid & 1) * 16;   // A staging: 128x32 tile
  const int rb = tid >> 3, cb = (tid & 7) * 16;   // B staging: 32x128 tile

  v8f acc[2][4];
#pragma unroll
  for (int i = 0; i < 2; ++i)
#pragma unroll
    for (int j = 0; j < 4; ++j)
#pragma unroll
      for (int e = 0; e < 8; ++e) acc[i][j][e] = 0.0f;

  for (int k0 = 0; k0 < K; k0 += BK) {
    // async-stage A (128x32) and B (32x128) tiles straight into LDS
    async_b128(&As[ra * BK + ca],     A + (size_t)(bm + ra) * K + k0 + ca);
    async_b128(&As[ra * BK + ca + 8], A + (size_t)(bm + ra) * K + k0 + ca + 8);
    async_b128(&Bs[rb * BN + cb],     Bm + (size_t)(k0 + rb) * Nc + bn + cb);
    async_b128(&Bs[rb * BN + cb + 8], Bm + (size_t)(k0 + rb) * Nc + bn + cb + 8);
    wait_async0();
    __syncthreads();

    // prefetch next K tiles into cache while we compute this one
    if (k0 + BK < K) {
      __builtin_prefetch(A + (size_t)(bm + ra) * K + k0 + BK + ca, 0, 1);
      __builtin_prefetch(Bm + (size_t)(k0 + BK + rb) * Nc + bn + cb, 0, 1);
    }

    v16h afrag[2], bfrag[4];
#pragma unroll
    for (int i = 0; i < 2; ++i) {
      int row = waveM * 32 + i * 16 + arow;
#pragma unroll
      for (int e = 0; e < 16; ++e) {
        int kk = (e >> 3) * 16 + kh * 8 + (e & 7);
        afrag[i][e] = As[row * BK + kk];
      }
    }
#pragma unroll
    for (int j = 0; j < 4; ++j) {
      int col = waveN * 64 + j * 16 + arow;
#pragma unroll
      for (int e = 0; e < 16; ++e) {
        int kk = (e >> 3) * 16 + kh * 8 + (e & 7);
        bfrag[j][e] = Bs[kk * BN + col];
      }
    }
#pragma unroll
    for (int i = 0; i < 2; ++i)
#pragma unroll
      for (int j = 0; j < 4; ++j)
        acc[i][j] = WMMA_F16(afrag[i], bfrag[j], acc[i][j]);
    __syncthreads();
  }

  const int g = lane >> 4;
#pragma unroll
  for (int i = 0; i < 2; ++i)
#pragma unroll
    for (int j = 0; j < 4; ++j) {
      int col = bn + waveN * 64 + j * 16 + (lane & 15);
      float bv = bias ? bias[col] : 0.0f;
#pragma unroll
      for (int r = 0; r < 8; ++r) {
        int row = bm + waveM * 32 + i * 16 + r + 8 * g;
        C[(size_t)row * Nc + col] = acc[i][j][r] + bv;
      }
    }
}

// -------------------------------------------- qk-layernorm + head split (f16)
// qkv f32 [B,N,3,H,D] -> Q (LN * D^-0.5), K (LN), V as f16 in [B,H,N,D]
__global__ __launch_bounds__(256) void qkv_ln_split(
    const float* __restrict__ qkv,
    const float* __restrict__ qg, const float* __restrict__ qb,
    const float* __restrict__ kg, const float* __restrict__ kb,
    _Float16* __restrict__ Q, _Float16* __restrict__ Kh,
    _Float16* __restrict__ Vh, int Bn, int Nn, int Hh) {
  const int D = 64;
  const int wave = threadIdx.x >> 5, lane = threadIdx.x & 31;
  long p = (long)blockIdx.x * 8 + wave;          // (b*N+n)*H + h
  int h = (int)(p % Hh);
  long bn = p / Hh;
  int n = (int)(bn % Nn);
  int b = (int)(bn / Nn);
  const int d0 = lane * 2;
  const float scale = 0.125f;                    // 64^-0.5

  size_t baseq = ((size_t)(bn * 3 + 0) * Hh + h) * D;
  size_t basek = ((size_t)(bn * 3 + 1) * Hh + h) * D;
  size_t basev = ((size_t)(bn * 3 + 2) * Hh + h) * D;
  size_t ob    = ((size_t)(b * Hh + h) * Nn + n) * D;

  { // Q layernorm
    float x0 = qkv[baseq + d0], x1 = qkv[baseq + d0 + 1];
    float s = x0 + x1;
#pragma unroll
    for (int m = 16; m >= 1; m >>= 1) s += __shfl_xor(s, m, 32);
    float mu = s * (1.0f / 64.0f);
    float e0 = x0 - mu, e1 = x1 - mu;
    float v = e0 * e0 + e1 * e1;
#pragma unroll
    for (int m = 16; m >= 1; m >>= 1) v += __shfl_xor(v, m, 32);
    float rstd = rsqrtf(v * (1.0f / 64.0f) + 1e-5f);
    Q[ob + d0]     = (_Float16)(((e0 * rstd) * qg[d0]     + qb[d0])     * scale);
    Q[ob + d0 + 1] = (_Float16)(((e1 * rstd) * qg[d0 + 1] + qb[d0 + 1]) * scale);
  }
  { // K layernorm
    float x0 = qkv[basek + d0], x1 = qkv[basek + d0 + 1];
    float s = x0 + x1;
#pragma unroll
    for (int m = 16; m >= 1; m >>= 1) s += __shfl_xor(s, m, 32);
    float mu = s * (1.0f / 64.0f);
    float e0 = x0 - mu, e1 = x1 - mu;
    float v = e0 * e0 + e1 * e1;
#pragma unroll
    for (int m = 16; m >= 1; m >>= 1) v += __shfl_xor(v, m, 32);
    float rstd = rsqrtf(v * (1.0f / 64.0f) + 1e-5f);
    Kh[ob + d0]     = (_Float16)((e0 * rstd) * kg[d0]     + kb[d0]);
    Kh[ob + d0 + 1] = (_Float16)((e1 * rstd) * kg[d0 + 1] + kb[d0 + 1]);
  }
  Vh[ob + d0]     = (_Float16)qkv[basev + d0];
  Vh[ob + d0 + 1] = (_Float16)qkv[basev + d0 + 1];
}

// ------------------------------------------------------------ flash attention
// Q,K,V f16 [B,H,N,64]; O f16 [B,N,H*64]; softmax(QK^T)V with online softmax.
__global__ __launch_bounds__(128) void flash_attn_wmma(
    const _Float16* __restrict__ Q, const _Float16* __restrict__ Kc,
    const _Float16* __restrict__ V, _Float16* __restrict__ O,
    int Bn, int Hh, int Nn) {
  const int D = 64;
  __shared__ __align__(16) _Float16 Ks[64 * 64];
  __shared__ __align__(16) _Float16 Vs[64 * 64];
  __shared__ __align__(16) _Float16 Ps[4][16 * 64];

  const int tid = threadIdx.x, lane = tid & 31, wave = tid >> 5;
  const int nblocks = Nn / 64;
  const int rbk = blockIdx.x % nblocks;
  const int bh = blockIdx.x / nblocks;
  const int h = bh % Hh, b = bh / Hh;
  const size_t headoff = (size_t)(b * Hh + h) * Nn * D;
  const int rowbase = rbk * 64 + wave * 16;
  const int arow = lane & 15, kh = lane >> 4, g = kh;
  const int sr = tid >> 1, sc = (tid & 1) * 32;   // K/V staging coords

  // Q A-fragments (K dim = D = 64 -> two k32 fragments), live whole kernel
  v16h qa[2];
#pragma unroll
  for (int f = 0; f < 2; ++f)
#pragma unroll
    for (int e = 0; e < 16; ++e) {
      int d = 32 * f + (e >> 3) * 16 + kh * 8 + (e & 7);
      qa[f][e] = Q[headoff + (size_t)(rowbase + arow) * D + d];
    }

  v8f o[4];
#pragma unroll
  for (int t = 0; t < 4; ++t)
#pragma unroll
    for (int e = 0; e < 8; ++e) o[t][e] = 0.0f;
  float mrow[8], lrow[8];
#pragma unroll
  for (int r = 0; r < 8; ++r) { mrow[r] = -1e30f; lrow[r] = 0.0f; }

  for (int kb = 0; kb < nblocks; ++kb) {
    { // async-stage K/V 64x64 tiles into LDS
      const _Float16* kg_ = Kc + headoff + (size_t)(kb * 64 + sr) * D + sc;
      const _Float16* vg_ = V  + headoff + (size_t)(kb * 64 + sr) * D + sc;
#pragma unroll
      for (int q4 = 0; q4 < 4; ++q4) {
        async_b128(&Ks[sr * 64 + sc + q4 * 8], kg_ + q4 * 8);
        async_b128(&Vs[sr * 64 + sc + q4 * 8], vg_ + q4 * 8);
      }
      wait_async0();
    }
    __syncthreads();

    if (kb + 1 < nblocks) { // prefetch next K/V block
      __builtin_prefetch(Kc + headoff + (size_t)((kb + 1) * 64 + sr) * D + sc, 0, 1);
      __builtin_prefetch(V  + headoff + (size_t)((kb + 1) * 64 + sr) * D + sc, 0, 1);
    }

    // S = q @ K^T : 4 key sub-tiles of 16, each two k32 WMMAs
    v8f s[4];
#pragma unroll
    for (int j = 0; j < 4; ++j) {
#pragma unroll
      for (int e = 0; e < 8; ++e) s[j][e] = 0.0f;
#pragma unroll
      for (int f = 0; f < 2; ++f) {
        v16h bf;
#pragma unroll
        for (int e = 0; e < 16; ++e) {
          int d = 32 * f + (e >> 3) * 16 + kh * 8 + (e & 7);
          bf[e] = Ks[(j * 16 + arow) * 64 + d];  // B[k=d][n=key]: col = lane&15
        }
        s[j] = WMMA_F16(qa[f], bf, s[j]);
      }
    }

    // online softmax: row max across 16-lane group + 4 tiles
    float mloc[8];
#pragma unroll
    for (int r = 0; r < 8; ++r) mloc[r] = -1e30f;
#pragma unroll
    for (int j = 0; j < 4; ++j)
#pragma unroll
      for (int r = 0; r < 8; ++r) mloc[r] = fmaxf(mloc[r], s[j][r]);
#pragma unroll
    for (int m = 8; m >= 1; m >>= 1)
#pragma unroll
      for (int r = 0; r < 8; ++r)
        mloc[r] = fmaxf(mloc[r], __shfl_xor(mloc[r], m, 32));

    float sc_[8], psum[8];
#pragma unroll
    for (int r = 0; r < 8; ++r) {
      float mn = fmaxf(mrow[r], mloc[r]);
      sc_[r] = __expf(mrow[r] - mn);
      mrow[r] = mn;
      psum[r] = 0.0f;
    }
#pragma unroll
    for (int j = 0; j < 4; ++j)
#pragma unroll
      for (int r = 0; r < 8; ++r) {
        float pv = __expf(s[j][r] - mrow[r]);
        s[j][r] = pv;
        psum[r] += pv;
      }
#pragma unroll
    for (int m = 8; m >= 1; m >>= 1)
#pragma unroll
      for (int r = 0; r < 8; ++r) psum[r] += __shfl_xor(psum[r], m, 32);
#pragma unroll
    for (int r = 0; r < 8; ++r) lrow[r] = lrow[r] * sc_[r] + psum[r];
#pragma unroll
    for (int t = 0; t < 4; ++t)
#pragma unroll
      for (int r = 0; r < 8; ++r) o[t][r] *= sc_[r];

    // C-layout P -> per-wave LDS -> A-layout fragments (f16)
#pragma unroll
    for (int j = 0; j < 4; ++j)
#pragma unroll
      for (int r = 0; r < 8; ++r)
        Ps[wave][(r + 8 * g) * 64 + j * 16 + arow] = (_Float16)s[j][r];

    v16h pa[2];
#pragma unroll
    for (int f = 0; f < 2; ++f)
#pragma unroll
      for (int e = 0; e < 16; ++e) {
        int kk = 32 * f + (e >> 3) * 16 + kh * 8 + (e & 7);
        pa[f][e] = Ps[wave][arow * 64 + kk];
      }

    // O += P @ V
#pragma unroll
    for (int t = 0; t < 4; ++t)
#pragma unroll
      for (int f = 0; f < 2; ++f) {
        v16h vb;
#pragma unroll
        for (int e = 0; e < 16; ++e) {
          int key = 32 * f + (e >> 3) * 16 + kh * 8 + (e & 7);
          vb[e] = Vs[key * 64 + t * 16 + arow];
        }
        o[t] = WMMA_F16(pa[f], vb, o[t]);
      }
    __syncthreads();
  }

  const int Cc = Hh * D;
#pragma unroll
  for (int t = 0; t < 4; ++t)
#pragma unroll
    for (int r = 0; r < 8; ++r) {
      int n = rowbase + r + 8 * g;
      int d = t * 16 + arow;
      O[(size_t)(b * Nn + n) * Cc + h * D + d] =
          (_Float16)(o[t][r] / lrow[r]);
    }
}

// --------------------------------------------------------------------- launch
extern "C" void kernel_launch(void* const* d_in, const int* in_sizes, int n_in,
                              void* d_out, int out_size, void* d_ws,
                              size_t ws_size, hipStream_t stream) {
  const float* x       = (const float*)d_in[0];
  const float* W_qkv   = (const float*)d_in[1];
  const float* q_gamma = (const float*)d_in[2];
  const float* q_beta  = (const float*)d_in[3];
  const float* k_gamma = (const float*)d_in[4];
  const float* k_beta  = (const float*)d_in[5];
  const float* W_proj  = (const float*)d_in[6];
  const float* b_proj  = (const float*)d_in[7];
  float* out = (float*)d_out;

  const int Bb = 2, Nn = 2048, Cc = 768, Hh = 12;
  const int BN_ = Bb * Nn;      // 4096
  const int C3 = 3 * Cc;        // 2304

  char* ws = (char*)d_ws;
  size_t off = 0;
  auto alloc = [&](size_t bytes) {
    void* p = ws + off;
    off += (bytes + 255) & ~(size_t)255;
    return p;
  };
  _Float16* x_h   = (_Float16*)alloc((size_t)BN_ * Cc * 2);
  _Float16* Wq_h  = (_Float16*)alloc((size_t)Cc * C3 * 2);
  _Float16* Wp_h  = (_Float16*)alloc((size_t)Cc * Cc * 2);
  float*    qkv32 = (float*)   alloc((size_t)BN_ * C3 * 4);
  _Float16* Qh    = (_Float16*)alloc((size_t)BN_ * Cc * 2);
  _Float16* Kh    = (_Float16*)alloc((size_t)BN_ * Cc * 2);
  _Float16* Vh    = (_Float16*)alloc((size_t)BN_ * Cc * 2);
  _Float16* AOh   = (_Float16*)alloc((size_t)BN_ * Cc * 2);

  { // f32 -> f16 conversions
    long n1 = (long)BN_ * Cc, n2 = (long)Cc * C3, n3 = (long)Cc * Cc;
    cvt_f32_to_f16<<<(int)((n1 + 255) / 256), 256, 0, stream>>>(x, x_h, n1);
    cvt_f32_to_f16<<<(int)((n2 + 255) / 256), 256, 0, stream>>>(W_qkv, Wq_h, n2);
    cvt_f32_to_f16<<<(int)((n3 + 255) / 256), 256, 0, stream>>>(W_proj, Wp_h, n3);
  }
  // qkv = x @ W_qkv   [4096 x 2304]
  gemm_f16_wmma<<<dim3(C3 / BN, BN_ / BM), 256, 0, stream>>>(
      x_h, Wq_h, nullptr, qkv32, BN_, C3, Cc);
  // qk-layernorm + split heads
  qkv_ln_split<<<(BN_ * Hh) / 8, 256, 0, stream>>>(
      qkv32, q_gamma, q_beta, k_gamma, k_beta, Qh, Kh, Vh, Bb, Nn, Hh);
  // flash attention
  flash_attn_wmma<<<Bb * Hh * (Nn / 64), 128, 0, stream>>>(
      Qh, Kh, Vh, AOh, Bb, Hh, Nn);
  // out = attn_out @ W_proj + b_proj
  gemm_f16_wmma<<<dim3(Cc / BN, BN_ / BM), 256, 0, stream>>>(
      AOh, Wp_h, b_proj, out, BN_, Cc, Cc);
}
